// BiRWKVLayer_65403761983502
// MI455X (gfx1250) — matrix operand order repaired and checked by reading, not compiled, not run on hardware
//
#include <hip/hip_runtime.h>

typedef __attribute__((ext_vector_type(16))) __bf16 v16bf;
typedef __attribute__((ext_vector_type(8)))  __bf16 v8bf;
typedef __attribute__((ext_vector_type(4)))  __bf16 v4bf;
typedef __attribute__((ext_vector_type(8)))  float  v8f;
typedef __attribute__((ext_vector_type(4)))  unsigned int u32x4;
typedef __attribute__((ext_vector_type(4)))  int          i32x4;
typedef __attribute__((ext_vector_type(8)))  int          i32x8;

#define BM 128
#define BN 128
#define BK 32   // == WMMA K for bf16

// ---------------------------------------------------------------------------
// Tensor Data Mover: DMA one 2D tile [tile_rows x tile_k] (bf16) from a
// row-major [rows_total x K] tensor into a contiguous row-major LDS tile.
// D# layout per CDNA5 ISA (group0: count/lds/global/type,
// group1: data_size=2B, tensor_dim0=K, tensor_dim1=rows, tile dims, stride0=K).
// TDM walks X (tile_k contiguous) then Y (tile_rows), LDS addr incrementing
// contiguously => exactly an LDS [tile_rows][tile_k] row-major fill.
// ---------------------------------------------------------------------------
__device__ __forceinline__ void tdm_load_tile_bf16(uint32_t lds_addr,
                                                   const void* gaddr,
                                                   int K, int rows_total,
                                                   int tile_rows, int tile_k) {
    const unsigned long long ga = (unsigned long long)(uintptr_t)gaddr;
    u32x4 g0;
    g0[0] = 1u;                                    // count=1, user descriptor
    g0[1] = lds_addr;                              // lds_addr [63:32]
    g0[2] = (unsigned)ga;                          // global_addr lo32
    g0[3] = (unsigned)(ga >> 32) | (2u << 30);     // global_addr hi25 | type=2
    i32x8 g1;
    g1[0] = (int)(1u << 16);                       // wg_mask=0, data_size=1 (2B)
    g1[1] = (int)(((unsigned)K & 0xFFFFu) << 16);  // tensor_dim0[15:0]
    g1[2] = (int)(((unsigned)K >> 16) | (((unsigned)rows_total & 0xFFFFu) << 16));
    g1[3] = (int)(((unsigned)rows_total >> 16) | ((unsigned)tile_k << 16));
    g1[4] = tile_rows;                             // tile_dim1 (tile_dim2=0)
    g1[5] = K;                                     // tensor_dim0_stride lo32
    g1[6] = 0;
    g1[7] = 0;
    i32x4 z4 = {0, 0, 0, 0};
#if __has_builtin(__builtin_amdgcn_tensor_load_to_lds)
#if __clang_major__ >= 23
    i32x8 z8 = {0, 0, 0, 0, 0, 0, 0, 0};
    __builtin_amdgcn_tensor_load_to_lds(g0, g1, z4, z4, z8, 0);
#else
    __builtin_amdgcn_tensor_load_to_lds(g0, g1, z4, z4, 0);
#endif
#else
    (void)z4;
    asm volatile("tensor_load_to_lds %0, %1" :: "s"(g0), "s"(g1) : "memory");
#endif
}

__device__ __forceinline__ void tdm_wait_all() {
    __builtin_amdgcn_s_wait_tensorcnt((short)0);
}

// ---------------------------------------------------------------------------
// f32 -> bf16 conversion (4 elements/thread, vectorized load/store)
// ---------------------------------------------------------------------------
__global__ void cvt_f32_bf16_kernel(const float* __restrict__ in,
                                    __bf16* __restrict__ out, int n4) {
    int i = blockIdx.x * blockDim.x + threadIdx.x;
    if (i >= n4) return;
    float4 v = reinterpret_cast<const float4*>(in)[i];
    v4bf o;
    o[0] = (__bf16)v.x; o[1] = (__bf16)v.y; o[2] = (__bf16)v.z; o[3] = (__bf16)v.w;
    reinterpret_cast<v4bf*>(out)[i] = o;
}

// ---------------------------------------------------------------------------
// Tiled bf16 WMMA GEMM, TDM-fed:
//   D[M,N] (f32) = A[M,K] (bf16, K-contig) x Bt[N,K] (bf16, K-contig)
// Block: 256 threads (8 waves), block tile 128x128, wave tile 32x64 (2x4 WMMA).
// Double-buffered LDS; wave 0 issues tensor_load_to_lds for tile kt+1 while
// all waves run WMMAs on tile kt; drain with s_wait_tensorcnt before barrier.
// ---------------------------------------------------------------------------
__global__ __launch_bounds__(256, 2)
void gemm_bf16_wmma_kernel(const __bf16* __restrict__ A,
                           const __bf16* __restrict__ Bt,
                           float* __restrict__ D,
                           int M, int N, int K) {
    __shared__ __bf16 As[2][BM][BK];
    __shared__ __bf16 Bs[2][BN][BK];

    const int tid  = threadIdx.x;
    const int lane = tid & 31;
    const int wave = tid >> 5;
    const int wm   = wave & 3;   // 4 waves along M
    const int wn   = wave >> 2;  // 2 waves along N
    const int block_m = blockIdx.y * BM;
    const int block_n = blockIdx.x * BN;

    // LDS aperture keeps the byte offset in addr[31:0] -> truncate flat addr.
    const uint32_t ldsA0 = (uint32_t)(uintptr_t)&As[0][0][0];
    const uint32_t ldsA1 = (uint32_t)(uintptr_t)&As[1][0][0];
    const uint32_t ldsB0 = (uint32_t)(uintptr_t)&Bs[0][0][0];
    const uint32_t ldsB1 = (uint32_t)(uintptr_t)&Bs[1][0][0];

    const __bf16* gA = A  + (size_t)block_m * K;   // tile row block, col kt*BK
    const __bf16* gB = Bt + (size_t)block_n * K;

    v8f acc[2][4];
#pragma unroll
    for (int mi = 0; mi < 2; ++mi)
#pragma unroll
        for (int ni = 0; ni < 4; ++ni)
#pragma unroll
            for (int e = 0; e < 8; ++e) acc[mi][ni][e] = 0.0f;

    const int KT = K / BK;

    if (wave == 0) {
        tdm_load_tile_bf16(ldsA0, gA, K, M, BM, BK);
        tdm_load_tile_bf16(ldsB0, gB, K, N, BN, BK);
        tdm_wait_all();
    }
    __syncthreads();

    // Fragment coordinates used here (contiguous k = 16h+s, not the ISA table order):
    //  A 16x32 bf16: lanes 0-15 -> M=lane,    K in {0..15}
    //                lanes16-31 -> M=lane-16, K in {16..31}
    //  B 32x16 bf16: lane holds column n=lane%16; 16 contiguous K,
    //                lanes 0-15 -> K=0..15, lanes 16-31 -> K=16..31
    const int a_row0 = wm * 32 + (lane & 15);
    const int a_kb   = (lane < 16) ? 0 : 16;
    const int b_col0 = wn * 64 + (lane & 15);
    const int b_kb   = (lane < 16) ? 0 : 16;

    for (int kt = 0; kt < KT; ++kt) {
        const int buf = kt & 1;
        const bool has_next = (kt + 1) < KT;

        if (has_next && wave == 0) {   // DMA next tile into the other buffer
            const uint32_t la = (buf == 0) ? ldsA1 : ldsA0;
            const uint32_t lb = (buf == 0) ? ldsB1 : ldsB0;
            tdm_load_tile_bf16(la, gA + (size_t)(kt + 1) * BK, K, M, BM, BK);
            tdm_load_tile_bf16(lb, gB + (size_t)(kt + 1) * BK, K, N, BN, BK);
        }

        v16bf afrag[2];
#pragma unroll
        for (int mi = 0; mi < 2; ++mi) {
            const v8bf lo = *reinterpret_cast<const v8bf*>(&As[buf][a_row0 + mi * 16][a_kb]);
            const v8bf hi = *reinterpret_cast<const v8bf*>(&As[buf][a_row0 + mi * 16][a_kb + 8]);
#pragma unroll
            for (int j = 0; j < 8; ++j) { afrag[mi][j] = lo[j]; afrag[mi][j + 8] = hi[j]; }
        }
        v16bf bfrag[4];
#pragma unroll
        for (int ni = 0; ni < 4; ++ni)
            bfrag[ni] = *reinterpret_cast<const v16bf*>(&Bs[buf][b_col0 + ni * 16][b_kb]);

#pragma unroll
        for (int mi = 0; mi < 2; ++mi)
#pragma unroll
            for (int ni = 0; ni < 4; ++ni)
                acc[mi][ni] = __builtin_amdgcn_wmma_f32_16x16x32_bf16(
                    false, afrag[mi], false, bfrag[ni],
                    (short)0, acc[mi][ni], false, false);

        if (has_next && wave == 0) tdm_wait_all();  // TDM overlapped with WMMAs
        __syncthreads();
    }

    // D layout (ISA 7.12.2): VGPR r, lane L -> (M = r + 8*(L>=16), N = L%16)
#pragma unroll
    for (int mi = 0; mi < 2; ++mi) {
#pragma unroll
        for (int ni = 0; ni < 4; ++ni) {
            const int nn  = block_n + wn * 64 + ni * 16 + (lane & 15);
            const int mm0 = block_m + wm * 32 + mi * 16 + ((lane >> 4) << 3);
            float* dp = D + (size_t)mm0 * N + nn;
#pragma unroll
            for (int r = 0; r < 8; ++r) dp[(size_t)r * N] = acc[mi][ni][r];
        }
    }
}

// ---------------------------------------------------------------------------
// Bidirectional WKV scan with running-max stabilization, fused sigmoid(r)*y.
// One thread per (dir, b, c) channel; coalesced over c; f32 math throughout.
// Writes cat[B,T,2C] in bf16 (the A operand of GEMM2).
// rkv layout: [B, T, 6, C] with d = dir*3C + {r,k,v}*C + c
// ---------------------------------------------------------------------------
__global__ __launch_bounds__(256, 4)
void wkv_scan_kernel(const float* __restrict__ rkv,
                     const float* __restrict__ td,  const float* __restrict__ tf,
                     const float* __restrict__ tdr, const float* __restrict__ tfr,
                     __bf16* __restrict__ cat, int Bq, int T, int C) {
    const int idx = blockIdx.x * blockDim.x + threadIdx.x;
    const int c   = idx % C;
    const int b   = (idx / C) % Bq;
    const int dir = idx / (C * Bq);

    const float w = -__expf(dir ? tdr[c] : td[c]);
    const float u =         dir ? tfr[c] : tf[c];

    float num = 0.0f, den = 0.0f, mx = -1e38f;

    const int strideT = 6 * C;
    const int t0 = dir ? (T - 1) : 0;
    const int dt = dir ? -1 : 1;
    const float* base = rkv + (size_t)b * T * strideT + dir * 3 * C + c;
    __bf16* outp = cat + (size_t)b * T * (2 * C) + dir * C + c;

    for (int s = 0; s < T; ++s) {
        const int t = t0 + dt * s;
        const float* pr = base + (size_t)t * strideT;
        if (s + 8 < T)  // global_prefetch_b8: pull future timestep toward L2/L0
            __builtin_prefetch(base + (size_t)(t + dt * 8) * strideT, 0, 0);

        const float rt = pr[0];
        const float kt = pr[C];
        const float vt = pr[2 * C];

        const float ku = kt + u;
        const float m  = fmaxf(mx, ku);
        const float e1 = __expf(mx - m);
        const float e2 = __expf(ku - m);
        const float y  = fmaf(e1, num, e2 * vt) / fmaf(e1, den, e2);

        const float mw  = mx + w;
        const float mn  = fmaxf(mw, kt);
        const float e1s = __expf(mw - mn);
        const float e2s = __expf(kt - mn);
        num = fmaf(e1s, num, e2s * vt);
        den = fmaf(e1s, den, e2s);
        mx  = mn;

        const float sg = 1.0f / (1.0f + __expf(-rt));
        outp[(size_t)t * (2 * C)] = (__bf16)(sg * y);
    }
}

// ---------------------------------------------------------------------------
// Launch: cvt(x,w1,w2) -> GEMM1 (TDM + wmma bf16) -> wkv scan -> GEMM2
// ---------------------------------------------------------------------------
extern "C" void kernel_launch(void* const* d_in, const int* in_sizes, int n_in,
                              void* d_out, int out_size, void* d_ws, size_t ws_size,
                              hipStream_t stream) {
    const float* x     = (const float*)d_in[0];  // [B,T,C]
    const float* rkv_w = (const float*)d_in[1];  // [6C, C]  (= Bt: N-major, K-contig)
    const float* out_w = (const float*)d_in[2];  // [C, 2C]  (= Bt: N-major, K-contig)
    const float* td    = (const float*)d_in[3];
    const float* tf    = (const float*)d_in[4];
    const float* tdr   = (const float*)d_in[5];
    const float* tfr   = (const float*)d_in[6];
    float* out = (float*)d_out;

    const int Bq = 4, T = 2048, C = 1024;
    const int M  = Bq * T;   // 8192
    const int N1 = 6 * C;    // 6144
    const int K1 = C;        // 1024
    const int N2 = C;        // 1024
    const int K2 = 2 * C;    // 2048

    char* p = (char*)d_ws;
    __bf16* xb  = (__bf16*)p;  p += (size_t)M  * K1 * sizeof(__bf16);
    __bf16* w1b = (__bf16*)p;  p += (size_t)N1 * K1 * sizeof(__bf16);
    __bf16* w2b = (__bf16*)p;  p += (size_t)N2 * K2 * sizeof(__bf16);
    float*  rkv = (float*)p;   p += (size_t)M  * N1 * sizeof(float);   // 192 MB ~= L2
    __bf16* cat = (__bf16*)p;  p += (size_t)M  * K2 * sizeof(__bf16);

    // bf16 casts (all counts divisible by 1024)
    {
        int n4 = (M * K1) / 4;
        cvt_f32_bf16_kernel<<<n4 / 256, 256, 0, stream>>>(x, xb, n4);
        n4 = (N1 * K1) / 4;
        cvt_f32_bf16_kernel<<<n4 / 256, 256, 0, stream>>>(rkv_w, w1b, n4);
        n4 = (N2 * K2) / 4;
        cvt_f32_bf16_kernel<<<n4 / 256, 256, 0, stream>>>(out_w, w2b, n4);
    }

    // GEMM1: rkv[M, 6C] = x @ rkv_w^T
    gemm_bf16_wmma_kernel<<<dim3(N1 / BN, M / BM), 256, 0, stream>>>(
        xb, w1b, rkv, M, N1, K1);

    // Bidirectional WKV scan (+ sigmoid gate), emits cat[M, 2C] bf16
    wkv_scan_kernel<<<(2 * Bq * C) / 256, 256, 0, stream>>>(
        rkv, td, tf, tdr, tfr, cat, Bq, T, C);

    // GEMM2: out[M, C] = cat @ out_w^T
    gemm_bf16_wmma_kernel<<<dim3(N2 / BN, M / BM), 256, 0, stream>>>(
        cat, w2b, out, M, N2, K2);
}
